// AttentionBlock_19421842112942
// MI455X (gfx1250) — compile-verified, matrix operand order checked by reading, not run on hardware
//
#include <hip/hip_runtime.h>
#include <stdint.h>

// ---------------- CDNA5 WMMA types ----------------
typedef __attribute__((ext_vector_type(16))) __bf16 v16bf;
typedef __attribute__((ext_vector_type(8)))  float  v8f;

union FragB16 {
  v16bf v;
  uint4 q[2];
  uint16_t h[16];
};

static __device__ __forceinline__ uint16_t f2bf(float f) {
  union { float f; uint32_t u; } c; c.f = f;
  uint32_t r = c.u + 0x7FFFu + ((c.u >> 16) & 1u);   // round-to-nearest-even
  return (uint16_t)(r >> 16);
}

static __device__ __forceinline__ v16bf load_frag16(const uint16_t* p0, const uint16_t* p1) {
  FragB16 f;
  f.q[0] = *(const uint4*)p0;   // global_load_b128
  f.q[1] = *(const uint4*)p1;   // global_load_b128
  return f.v;
}

static __device__ __forceinline__ v8f wmma_bf16(v16bf a, v16bf b, v8f c) {
  // D = A(16x32 bf16) * B(32x16 bf16) + C(16x16 f32)
  return __builtin_amdgcn_wmma_f32_16x16x32_bf16(false, a, false, b, (short)0, c,
                                                 false, false);
}

static __device__ __forceinline__ v8f v8f_zero() {
  v8f z = {0.f, 0.f, 0.f, 0.f, 0.f, 0.f, 0.f, 0.f};
  return z;
}

// Problem constants (B=4, H=W=64, C=512, Cr=64)
#define NTOK   16384   // B * H * W
#define NB     4096    // tokens per batch
#define CDIM   512
#define DDIM   64

// ---------------- Kernel 0: weight transpose+convert ----------------
// Wt[n, c] = [Wf | Wg | Wh][c, n]  as bf16, n in [0,640), c in [0,512)
__global__ void k_wt(const float* __restrict__ Wf, const float* __restrict__ Wg,
                     const float* __restrict__ Wh, uint16_t* __restrict__ Wt) {
  int idx = blockIdx.x * blockDim.x + threadIdx.x;   // 640*512 = 327680
  int n = idx >> 9;
  int c = idx & 511;
  float v;
  if (n < 64)       v = Wf[c * 64 + n];
  else if (n < 128) v = Wg[c * 64 + (n - 64)];
  else              v = Wh[c * 512 + (n - 128)];
  Wt[(size_t)n * 512 + c] = f2bf(v);
}

// ---------------- Kernel 1: fused qkv projection GEMM ----------------
// C[16384, 640] = x[16384,512] * W[512,640]; outputs split into
// q[tok,64], k[tok,64] (bf16) and vT[512,16384] (bf16, transposed).
__global__ void __launch_bounds__(128) k_qkv(
    const float* __restrict__ x, const uint16_t* __restrict__ Wt,
    uint16_t* __restrict__ qb, uint16_t* __restrict__ kb,
    uint16_t* __restrict__ vT) {
  const int lane = threadIdx.x & 31;
  const int w    = threadIdx.x >> 5;
  const int m    = lane & 15;
  const int half = lane >> 4;
  const int t0   = blockIdx.x * 16;          // token row tile
  const int ct   = blockIdx.y * 4 + w;       // output column tile (0..39)

  v8f acc = v8f_zero();
  const float*    xrow = x  + (size_t)(t0 + m) * CDIM;
  const uint16_t* wrow = Wt + (size_t)(ct * 16 + m) * CDIM;

  for (int k0 = 0; k0 < CDIM; k0 += 32) {
    // A fragment: x rows (fp32 -> bf16), K = k0 + half*8 + {0..7, 16..23}
    FragB16 a;
    const float* ap = xrow + k0 + half * 8;
    float4 f0 = *(const float4*)(ap);
    float4 f1 = *(const float4*)(ap + 4);
    float4 f2 = *(const float4*)(ap + 16);
    float4 f3 = *(const float4*)(ap + 20);
    a.h[0]  = f2bf(f0.x); a.h[1]  = f2bf(f0.y); a.h[2]  = f2bf(f0.z); a.h[3]  = f2bf(f0.w);
    a.h[4]  = f2bf(f1.x); a.h[5]  = f2bf(f1.y); a.h[6]  = f2bf(f1.z); a.h[7]  = f2bf(f1.w);
    a.h[8]  = f2bf(f2.x); a.h[9]  = f2bf(f2.y); a.h[10] = f2bf(f2.z); a.h[11] = f2bf(f2.w);
    a.h[12] = f2bf(f3.x); a.h[13] = f2bf(f3.y); a.h[14] = f2bf(f3.z); a.h[15] = f2bf(f3.w);
    // B fragment: Wt row (K = k0 + half*8 + {0..7, 16..23}, same as A)
    v16bf bf = load_frag16(wrow + k0 + half * 8, wrow + k0 + half * 8 + 16);
    acc = wmma_bf16(a.v, bf, acc);
  }

  // C layout: VGPR i, lanes0-15 -> row i, lanes16-31 -> row i+8; col = lane&15
  const int co = ct * 16 + m;
#pragma unroll
  for (int i = 0; i < 8; ++i) {
    int row = t0 + i + 8 * half;
    uint16_t hv = f2bf(acc[i]);
    if (ct < 4)      qb[(size_t)row * DDIM + co] = hv;
    else if (ct < 8) kb[(size_t)row * DDIM + (co - 64)] = hv;
    else             vT[(size_t)(co - 128) * NTOK + row] = hv;
  }
}

// ---------------- Kernel 2: flash attention + epilogue ----------------
// Block = 32 query rows of one batch, 8 waves (256 thr).
// Wave = (row tile w&1) x (128 channels at (w>>1)*128).
// K chunks (32 keys, 4KB) staged into LDS with async-to-LDS, double buffered.
__global__ void __launch_bounds__(256) k_attn(
    const uint16_t* __restrict__ qb, const uint16_t* __restrict__ kb,
    const uint16_t* __restrict__ vT, const float* __restrict__ x,
    const float* __restrict__ gamma, float* __restrict__ out) {
  __shared__ __align__(16) uint16_t klds[2][32 * DDIM];   // 2 x 4KB K chunk
  __shared__ __align__(16) uint16_t pbuf[8][16 * 32];     // per-wave P staging

  const int tid  = threadIdx.x;
  const int lane = tid & 31;
  const int w    = tid >> 5;
  const int m    = lane & 15;
  const int half = lane >> 4;
  const int rt   = w & 1;            // row tile within block (0/1)
  const int cb0  = (w >> 1) * 128;   // wave's channel base
  const int b    = blockIdx.y;
  const int tb   = b * NB + blockIdx.x * 32;   // global token base of Q block

  // Q A-fragments for this wave's 16 rows (K = d in [0,32), [32,64))
  const uint16_t* qrow = qb + (size_t)(tb + rt * 16 + m) * DDIM;
  const v16bf qf0 = load_frag16(qrow + half * 8,      qrow + half * 8 + 16);
  const v16bf qf1 = load_frag16(qrow + 32 + half * 8, qrow + 32 + half * 8 + 16);

  v8f acc[8];
#pragma unroll
  for (int t = 0; t < 8; ++t) acc[t] = v8f_zero();
  float mi[8], li[8];
#pragma unroll
  for (int i = 0; i < 8; ++i) { mi[i] = -1e30f; li[i] = 0.f; }

  const uint16_t* kbb = kb + (size_t)b * NB * DDIM;
  const uint16_t* vbb = vT + (size_t)b * NB;

  // Each thread async-copies 16B of the 4KB chunk: byte off = tid*16.
  const uint32_t kgoff = (uint32_t)tid * 16u;               // byte offset in chunk
  uint32_t ldsdst[2];
  ldsdst[0] = (uint32_t)(size_t)&klds[0][tid * 8];
  ldsdst[1] = (uint32_t)(size_t)&klds[1][tid * 8];

  // prologue: stage chunk 0 into buffer 0
  asm volatile("global_load_async_to_lds_b128 %0, %1, %2"
               :: "v"(ldsdst[0]), "v"(kgoff), "s"(kbb) : "memory");

#pragma unroll 1
  for (int n0 = 0; n0 < NB; n0 += 32) {
    const int nb = (n0 >> 5) & 1;
    if (n0 + 32 < NB) {
      // stage next chunk into the other buffer (freed by last iter's barrier)
      uint32_t go = kgoff + (uint32_t)(n0 + 32) * (DDIM * 2);
      asm volatile("global_load_async_to_lds_b128 %0, %1, %2"
                   :: "v"(ldsdst[nb ^ 1]), "v"(go), "s"(kbb) : "memory");
      asm volatile("s_wait_asynccnt 0x1" ::: "memory");  // this chunk's copy done
      __builtin_prefetch(vbb + (size_t)(cb0 + m) * NTOK + n0 + 32, 0, 1);
    } else {
      asm volatile("s_wait_asynccnt 0x0" ::: "memory");
    }
    __syncthreads();   // all waves' slices of chunk n0 are in klds[nb]

    // ---- S = Q K^T for 32 keys -> two 16x16 f32 tiles (B frags from LDS) ----
    v8f s0 = v8f_zero(), s1 = v8f_zero();
    {
      const uint16_t* kr0 = &klds[nb][(m) * DDIM + half * 8];
      s0 = wmma_bf16(qf0, load_frag16(kr0,      kr0 + 16), s0);
      s0 = wmma_bf16(qf1, load_frag16(kr0 + 32, kr0 + 48), s0);
      const uint16_t* kr1 = &klds[nb][(16 + m) * DDIM + half * 8];
      s1 = wmma_bf16(qf0, load_frag16(kr1,      kr1 + 16), s1);
      s1 = wmma_bf16(qf1, load_frag16(kr1 + 32, kr1 + 48), s1);
    }

    // ---- online softmax (row stats via 16-lane shfl groups) ----
#pragma unroll
    for (int i = 0; i < 8; ++i) {
      float mx = fmaxf(s0[i], s1[i]);
#pragma unroll
      for (int o = 1; o < 16; o <<= 1) mx = fmaxf(mx, __shfl_xor(mx, o, 16));
      float mnew  = fmaxf(mi[i], mx);
      float alpha = __expf(mi[i] - mnew);
      float p0 = __expf(s0[i] - mnew);
      float p1 = __expf(s1[i] - mnew);
      float rs = p0 + p1;
#pragma unroll
      for (int o = 1; o < 16; o <<= 1) rs += __shfl_xor(rs, o, 16);
      li[i] = li[i] * alpha + rs;
      mi[i] = mnew;
#pragma unroll
      for (int t = 0; t < 8; ++t) acc[t][i] *= alpha;
      int r = i + 8 * half;
      pbuf[w][r * 32 + m]      = f2bf(p0);   // ds_store_b16
      pbuf[w][r * 32 + 16 + m] = f2bf(p1);
    }

    // C-layout -> A-layout transpose through per-wave LDS (in-order DS per wave)
    asm volatile("s_wait_dscnt 0x0" ::: "memory");
    FragB16 pf;
    pf.q[0] = *(const uint4*)(&pbuf[w][m * 32 + half * 8]);        // ds_load_b128
    pf.q[1] = *(const uint4*)(&pbuf[w][m * 32 + half * 8 + 16]);

    // ---- acc += P V : 8 channel tiles of 16 (B frags from global vT) ----
    const uint16_t* vbase = vbb + n0 + half * 8;
#pragma unroll
    for (int t = 0; t < 8; ++t) {
      const uint16_t* vr = vbase + (size_t)(cb0 + t * 16 + m) * NTOK;
      acc[t] = wmma_bf16(pf.v, load_frag16(vr, vr + 16), acc[t]);
    }

    __syncthreads();   // everyone done reading klds[nb] before it is refilled
  }

  // ---- epilogue: out = gamma * (acc / l) + x ----
  const float g = gamma[0];
#pragma unroll
  for (int t = 0; t < 8; ++t) {
    int col = cb0 + t * 16 + m;
#pragma unroll
    for (int i = 0; i < 8; ++i) {
      int row = tb + rt * 16 + i + 8 * half;
      size_t oi = (size_t)row * CDIM + col;
      out[oi] = g * (acc[t][i] / li[i]) + x[oi];
    }
  }
}

// ---------------- launch ----------------
extern "C" void kernel_launch(void* const* d_in, const int* in_sizes, int n_in,
                              void* d_out, int out_size, void* d_ws, size_t ws_size,
                              hipStream_t stream) {
  (void)in_sizes; (void)n_in; (void)out_size; (void)ws_size;
  const float* x     = (const float*)d_in[0];
  const float* Wf    = (const float*)d_in[1];
  const float* Wg    = (const float*)d_in[2];
  const float* Wh    = (const float*)d_in[3];
  const float* gamma = (const float*)d_in[4];
  float* out = (float*)d_out;

  char* ws = (char*)d_ws;
  uint16_t* qb = (uint16_t*)(ws);                                    //  2 MB
  uint16_t* kb = (uint16_t*)(ws + (size_t)2  * 1024 * 1024);         //  2 MB
  uint16_t* vT = (uint16_t*)(ws + (size_t)4  * 1024 * 1024);         // 16 MB
  uint16_t* Wt = (uint16_t*)(ws + (size_t)20 * 1024 * 1024);         // 640 KB

  k_wt  <<<dim3(1280),     dim3(256), 0, stream>>>(Wf, Wg, Wh, Wt);
  k_qkv <<<dim3(1024, 10), dim3(128), 0, stream>>>(x, Wt, qb, kb, vT);
  k_attn<<<dim3(128, 4),   dim3(256), 0, stream>>>(qb, kb, vT, x, gamma, out);
}
